// DDG_Net_nogcn_43834436223251
// MI455X (gfx1250) — compile-verified, matrix operand order checked by reading, not run on hardware
//
#include <hip/hip_runtime.h>
#include <hip/hip_bf16.h>

// ---------------------------------------------------------------------------
// CDNA5 (gfx1250): GEMM-shaped work (k=3 convs, T x T similarity, feature
// propagation) on v_wmma_f32_16x16x32_f16. Each wave32 owns a 32x32 output
// tile (2x2 WMMA register tiling -> 4 wmma per K-step, 8 in the fused
// similarity kernel). A-tiles staged row-major ([m][k], contiguous ->
// ds_store_b128 / ds_load_b128); B-tiles staged transposed ([n][k]) so the
// B fragment has identical per-lane addressing as A: two ds_load_b128.
// ---------------------------------------------------------------------------

typedef __attribute__((ext_vector_type(16))) _Float16 v16h;
typedef __attribute__((ext_vector_type(8)))  float    v8f;

#define EPSV 1e-12f
#define WMMA_F16(a, b, c) \
    __builtin_amdgcn_wmma_f32_16x16x32_f16(false, (a), false, (b), (short)0, (c), false, false)

// Fragment loader for a 16x32 tile stored row-major with stride 32 halfs.
// Per CDNA5 ISA 7.12.2 (16-bit A 16x32): lane m = L%16, g = L/16; the lane's
// 16 halfs are two contiguous 8-half runs at [m][g*8] and [m][16+g*8].
// The transposed-B layout makes B fragments use the same mapping (m -> n).
__device__ __forceinline__ v16h frag16x32(const _Float16* __restrict__ P) {
    const int lane = threadIdx.x & 31;
    const int m = lane & 15, g = lane >> 4;
    const _Float16* p0 = P + m * 32 + g * 8;
    v16h r;
#pragma unroll
    for (int i = 0; i < 8; ++i) {
        r[i]     = p0[i];       // K = g*8 + i        (VGPRs 0..3)
        r[8 + i] = p0[16 + i];  // K = 16 + g*8 + i   (VGPRs 4..7)
    }
    return r;
}

// ---------------------------------------------------------------------------
// conv1d k=3 pad=1 + bias + leaky_relu(0.2). out[m,n] = sum_k W[m,k]*im2col.
// W is [Cout, Cin*3] contiguous -> A tiles are plain rows of W.
// One wave per 32x32 (Cout x T) tile; K = Cin*3 (multiple of 32).
// ---------------------------------------------------------------------------
__global__ void __launch_bounds__(32)
conv_k3_wmma(const float* __restrict__ x, const float* __restrict__ W,
             const float* __restrict__ bias, float* __restrict__ y,
             int Cin, int Cout, int T) {
    __shared__ _Float16 As[32 * 32];   // [m][k]
    __shared__ _Float16 Bt[32 * 32];   // [n][k] (transposed)
    const int lane = threadIdx.x;
    const int n0 = blockIdx.x * 32;
    const int m0 = blockIdx.y * 32;
    const int bb = blockIdx.z;
    const float* xb = x + (size_t)bb * Cin * T;
    float* yb = y + (size_t)bb * Cout * T;
    const int K = Cin * 3;
    v8f acc[2][2] = {};

    for (int k0 = 0; k0 < K; k0 += 32) {
        // A tile: lane = m row, 32 contiguous weights
        const float* wrow = W + (size_t)(m0 + lane) * K + k0;
        __builtin_prefetch((const void*)(wrow + 32), 0, 0);
#pragma unroll
        for (int j = 0; j < 32; ++j)
            As[lane * 32 + j] = (_Float16)wrow[j];
        // B tile: lane = k row of im2col; scatter into Bt[n][k]
        {
            const int k = k0 + lane;
            const int ci = k / 3;
            const int rr = k - ci * 3;
            const float* xr = xb + (size_t)ci * T;
            const int tb = n0 + rr - 1;
#pragma unroll
            for (int j = 0; j < 32; ++j) {
                const int t = tb + j;
                Bt[j * 32 + lane] =
                    (t >= 0 && t < T) ? (_Float16)xr[t] : (_Float16)0.f;
            }
        }
        __syncthreads();
        v16h a0 = frag16x32(As);
        v16h a1 = frag16x32(As + 16 * 32);
        v16h b0 = frag16x32(Bt);
        v16h b1 = frag16x32(Bt + 16 * 32);
        acc[0][0] = WMMA_F16(a0, b0, acc[0][0]);
        acc[0][1] = WMMA_F16(a0, b1, acc[0][1]);
        acc[1][0] = WMMA_F16(a1, b0, acc[1][0]);
        acc[1][1] = WMMA_F16(a1, b1, acc[1][1]);
        __syncthreads();
    }

    const int n = lane & 15, g = lane >> 4;
#pragma unroll
    for (int mt = 0; mt < 2; ++mt)
#pragma unroll
        for (int nt = 0; nt < 2; ++nt)
#pragma unroll
            for (int v = 0; v < 8; ++v) {
                const int m = m0 + mt * 16 + v + 8 * g;
                const int nn = n0 + nt * 16 + n;
                if (nn < T) {
                    float val = acc[mt][nt][v] + bias[m];
                    val = (val > 0.f) ? val : 0.2f * val;   // leaky_relu 0.2
                    yb[(size_t)m * T + nn] = val;
                }
            }
}

// ---------------------------------------------------------------------------
// 1x1 conv (Cout=1) + sigmoid
// ---------------------------------------------------------------------------
__global__ void conv1_sigmoid(const float* __restrict__ h,
                              const float* __restrict__ w,
                              const float* __restrict__ b0,
                              float* __restrict__ atn, int C, int T) {
    const int t = blockIdx.x * blockDim.x + threadIdx.x;
    const int bb = blockIdx.y;
    if (t >= T) return;
    const float* hb = h + (size_t)bb * C * T;
    float s = b0[0];
    for (int c = 0; c < C; ++c) s += w[c] * hb[(size_t)c * T + t];
    atn[(size_t)bb * T + t] = 1.f / (1.f + expf(-s));
}

// ---------------------------------------------------------------------------
// 1 / max(||x[b,:,t]||2, eps)
// ---------------------------------------------------------------------------
__global__ void invnorm_kernel(const float* __restrict__ x,
                               float* __restrict__ inv, int D, int T) {
    const int t = blockIdx.x * blockDim.x + threadIdx.x;
    const int bb = blockIdx.y;
    if (t >= T) return;
    const float* xb = x + (size_t)bb * D * T;
    float s = 0.f;
    for (int d = 0; d < D; ++d) {
        const float v = xb[(size_t)d * T + t];
        s += v * v;
    }
    inv[(size_t)bb * T + t] = 1.f / fmaxf(sqrtf(s), EPSV);
}

// ---------------------------------------------------------------------------
// fusion[b,t,u] = 0.5*(vn.vn + fn.fn), zeroed below 0.8.
// One wave per 32x32 (t x u) tile; v and f share staging barriers:
// 8 wmma per K-step. Normalization applied during fp32->f16 staging.
// Both operands need a transpose (A[m=t][k=d], B[n=u][k=d] from [d][t]
// global layout), so both are scatter-staged from contiguous global rows.
// ---------------------------------------------------------------------------
__global__ void __launch_bounds__(32)
sim_fusion_wmma(const float* __restrict__ vfeat, const float* __restrict__ ffeat,
                const float* __restrict__ invv, const float* __restrict__ invf,
                float* __restrict__ fusion, int D, int T) {
    __shared__ _Float16 Av[32 * 32], Bv[32 * 32], Af[32 * 32], Bf[32 * 32];
    __shared__ float ivt[32], ivu[32], ift[32], ifu[32];
    const int lane = threadIdx.x;
    const int u0 = blockIdx.x * 32;
    const int t0 = blockIdx.y * 32;
    const int bb = blockIdx.z;
    const float* vb = vfeat + (size_t)bb * D * T;
    const float* fb = ffeat + (size_t)bb * D * T;
    {
        const int t = t0 + lane, u = u0 + lane;
        ivt[lane] = (t < T) ? invv[(size_t)bb * T + t] : 0.f;
        ivu[lane] = (u < T) ? invv[(size_t)bb * T + u] : 0.f;
        ift[lane] = (t < T) ? invf[(size_t)bb * T + t] : 0.f;
        ifu[lane] = (u < T) ? invf[(size_t)bb * T + u] : 0.f;
    }
    __syncthreads();

    v8f aV[2][2] = {}, aF[2][2] = {};
    for (int k0 = 0; k0 < D; k0 += 32) {
        const float* vrow = vb + (size_t)(k0 + lane) * T;
        const float* frow = fb + (size_t)(k0 + lane) * T;
#pragma unroll
        for (int j = 0; j < 32; ++j) {
            const int t = t0 + j, u = u0 + j;
            Av[j * 32 + lane] = (_Float16)((t < T) ? vrow[t] * ivt[j] : 0.f);
            Bv[j * 32 + lane] = (_Float16)((u < T) ? vrow[u] * ivu[j] : 0.f);
            Af[j * 32 + lane] = (_Float16)((t < T) ? frow[t] * ift[j] : 0.f);
            Bf[j * 32 + lane] = (_Float16)((u < T) ? frow[u] * ifu[j] : 0.f);
        }
        __syncthreads();
        {
            v16h a0 = frag16x32(Av), a1 = frag16x32(Av + 512);
            v16h b0 = frag16x32(Bv), b1 = frag16x32(Bv + 512);
            aV[0][0] = WMMA_F16(a0, b0, aV[0][0]);
            aV[0][1] = WMMA_F16(a0, b1, aV[0][1]);
            aV[1][0] = WMMA_F16(a1, b0, aV[1][0]);
            aV[1][1] = WMMA_F16(a1, b1, aV[1][1]);
        }
        {
            v16h a0 = frag16x32(Af), a1 = frag16x32(Af + 512);
            v16h b0 = frag16x32(Bf), b1 = frag16x32(Bf + 512);
            aF[0][0] = WMMA_F16(a0, b0, aF[0][0]);
            aF[0][1] = WMMA_F16(a0, b1, aF[0][1]);
            aF[1][0] = WMMA_F16(a1, b0, aF[1][0]);
            aF[1][1] = WMMA_F16(a1, b1, aF[1][1]);
        }
        __syncthreads();
    }

    const int n = lane & 15, g = lane >> 4;
#pragma unroll
    for (int mt = 0; mt < 2; ++mt)
#pragma unroll
        for (int nt = 0; nt < 2; ++nt)
#pragma unroll
            for (int v = 0; v < 8; ++v) {
                const int t = t0 + mt * 16 + v + 8 * g;
                const int u = u0 + nt * 16 + n;
                if (t < T && u < T) {
                    float fus = 0.5f * (aV[mt][nt][v] + aF[mt][nt][v]);
                    if (fus < 0.8f) fus = 0.f;   // SIMILARITY_TH
                    fusion[((size_t)bb * T + t) * T + u] = fus;
                }
            }
}

// ---------------------------------------------------------------------------
// a_row / b_row indicators
// ---------------------------------------------------------------------------
__global__ void rowmask_kernel(const float* __restrict__ va,
                               const float* __restrict__ fa,
                               float* __restrict__ arow,
                               float* __restrict__ brow, int n) {
    const int i = blockIdx.x * blockDim.x + threadIdx.x;
    if (i >= n) return;
    const float v = va[i], f = fa[i];
    arow[i] = (v >= 0.6f && f >= 0.6f) ? 1.f : 0.f;
    brow[i] = (v < 0.4f && f < 0.4f) ? 1.f : 0.f;
}

__device__ __forceinline__ float block_reduce256(float v, float* red) {
    const int tid = threadIdx.x;
    red[tid] = v;
    __syncthreads();
    for (int s = 128; s > 0; s >>= 1) {
        if (tid < s) red[tid] += red[tid + s];
        __syncthreads();
    }
    const float r = red[0];
    __syncthreads();
    return r;
}

// ---------------------------------------------------------------------------
// Per-column (b,u): top-k threshold (values are 0 or >=0.8 so the k-th
// largest is 0 unless more than k entries survived the 0.8 cut; binary
// search only in that rare case), then the three L1 column sums.
// ---------------------------------------------------------------------------
__global__ void __launch_bounds__(256)
colstats_kernel(const float* __restrict__ fusion,
                const float* __restrict__ arow, const float* __restrict__ brow,
                float* __restrict__ thr, float* __restrict__ sa,
                float* __restrict__ sb, float* __restrict__ samb,
                int T, int K) {
    __shared__ float red[256];
    const int u = blockIdx.x;
    const int bb = blockIdx.y;
    const int tid = threadIdx.x;
    const float* col = fusion + ((size_t)bb * T) * T + u;   // element t: col[t*T]
    const float* ar = arow + (size_t)bb * T;
    const float* br = brow + (size_t)bb * T;

    float cnt = 0.f;
    for (int t = tid; t < T; t += 256)
        if (col[(size_t)t * T] > 0.f) cnt += 1.f;
    const int total = (int)block_reduce256(cnt, red);

    float th = 0.f;
    if (total > K) {
        float lo = 0.8f, hi = 1.0f;
        for (int it = 0; it < 30; ++it) {
            const float mid = 0.5f * (lo + hi);
            float c = 0.f;
            for (int t = tid; t < T; t += 256)
                if (col[(size_t)t * T] >= mid) c += 1.f;
            const int cc = (int)block_reduce256(c, red);
            if (cc > K) lo = mid; else hi = mid;
        }
        th = hi;
    }

    float la = 0.f, lb = 0.f, lm = 0.f;
    for (int t = tid; t < T; t += 256) {
        const float f = col[(size_t)t * T];
        const float kept = (f >= th) ? f : 0.f;
        la += ar[t] * kept;
        lb += br[t] * kept;
        const float ab = ar[t] + br[t];
        lm += ((ab != 0.f) || (t == u)) ? kept : 0.f;
    }
    la = block_reduce256(la, red);
    lb = block_reduce256(lb, red);
    lm = block_reduce256(lm, red);
    if (tid == 0) {
        const size_t o = (size_t)bb * T + u;
        thr[o] = th;
        sa[o] = la;
        sb[o] = lb;
        samb[o] = lm;
    }
}

// ---------------------------------------------------------------------------
// A = l1n(adj_a)+l1n(adj_b)+l1n(adj_amb), built in place over fusion
// ---------------------------------------------------------------------------
__global__ void buildA_kernel(float* __restrict__ fusion,
                              const float* __restrict__ arow,
                              const float* __restrict__ brow,
                              const float* __restrict__ thr,
                              const float* __restrict__ sa,
                              const float* __restrict__ sb,
                              const float* __restrict__ samb, int T) {
    const int bb = blockIdx.y;
    const size_t idx = (size_t)blockIdx.x * blockDim.x + threadIdx.x;
    if (idx >= (size_t)T * T) return;
    const int t = (int)(idx / T);
    const int u = (int)(idx - (size_t)t * T);
    const size_t off = (size_t)bb * T * T + idx;
    const size_t rt = (size_t)bb * T + t;
    const size_t ru = (size_t)bb * T + u;

    const float f = fusion[off];
    const float kept = (f >= thr[ru]) ? f : 0.f;
    const float at = arow[rt], au = arow[ru];
    const float bt = brow[rt], bu = brow[ru];
    const float abt = at + bt, abu = au + bu;

    float res = 0.f;
    if (at != 0.f && au != 0.f) res += kept / fmaxf(sa[ru], EPSV);
    if (bt != 0.f && bu != 0.f) res += kept / fmaxf(sb[ru], EPSV);
    if (abu == 0.f && (abt != 0.f || t == u)) res += kept / fmaxf(samb[ru], EPSV);
    fusion[off] = res;
}

// ---------------------------------------------------------------------------
// new_x = 0.5*(x + x @ A), [D,T] x [T,T] per batch. K = T = 2000 ->
// zero-padded staging for the 16-wide remainder; 32x32 per wave.
// ---------------------------------------------------------------------------
__global__ void __launch_bounds__(32)
prop_wmma(const float* __restrict__ x, const float* __restrict__ Amat,
          float* __restrict__ out, int D, int T) {
    __shared__ _Float16 As[32 * 32];   // [m=d][k=t]
    __shared__ _Float16 Bt[32 * 32];   // [n=u][k=t]
    const int lane = threadIdx.x;
    const int u0 = blockIdx.x * 32;
    const int d0 = blockIdx.y * 32;
    const int bb = blockIdx.z;
    const float* xb = x + (size_t)bb * D * T;
    const float* Ab = Amat + (size_t)bb * T * T;
    float* ob = out + (size_t)bb * D * T;
    v8f acc[2][2] = {};

    for (int k0 = 0; k0 < T; k0 += 32) {
        const float* xrow = xb + (size_t)(d0 + lane) * T + k0;
        __builtin_prefetch((const void*)(xrow + 32), 0, 0);
#pragma unroll
        for (int j = 0; j < 32; ++j)
            As[lane * 32 + j] =
                (k0 + j < T) ? (_Float16)xrow[j] : (_Float16)0.f;
        const int k = k0 + lane;
        if (k < T) {
            const float* Ar = Ab + (size_t)k * T;
#pragma unroll
            for (int j = 0; j < 32; ++j) {
                const int u = u0 + j;
                Bt[j * 32 + lane] = (u < T) ? (_Float16)Ar[u] : (_Float16)0.f;
            }
        } else {
#pragma unroll
            for (int j = 0; j < 32; ++j) Bt[j * 32 + lane] = (_Float16)0.f;
        }
        __syncthreads();
        v16h a0 = frag16x32(As);
        v16h a1 = frag16x32(As + 16 * 32);
        v16h b0 = frag16x32(Bt);
        v16h b1 = frag16x32(Bt + 16 * 32);
        acc[0][0] = WMMA_F16(a0, b0, acc[0][0]);
        acc[0][1] = WMMA_F16(a0, b1, acc[0][1]);
        acc[1][0] = WMMA_F16(a1, b0, acc[1][0]);
        acc[1][1] = WMMA_F16(a1, b1, acc[1][1]);
        __syncthreads();
    }

    const int n = lane & 15, g = lane >> 4;
#pragma unroll
    for (int mt = 0; mt < 2; ++mt)
#pragma unroll
        for (int nt = 0; nt < 2; ++nt)
#pragma unroll
            for (int v = 0; v < 8; ++v) {
                const int d = d0 + mt * 16 + v + 8 * g;
                const int u = u0 + nt * 16 + n;
                if (u < T) {
                    const size_t o = (size_t)d * T + u;
                    ob[o] = 0.5f * (xb[o] + acc[mt][nt][v]);
                }
            }
}

// ---------------------------------------------------------------------------
static void run_attention(const float* x, const float* w1, const float* b1,
                          const float* w2, const float* b2, const float* w3,
                          const float* b3, float* h1, float* h2, float* atn,
                          int B, int D, int C, int T, hipStream_t stream) {
    const int NT = (T + 31) / 32;
    conv_k3_wmma<<<dim3(NT, C / 32, B), 32, 0, stream>>>(x, w1, b1, h1, D, C, T);
    conv_k3_wmma<<<dim3(NT, C / 32, B), 32, 0, stream>>>(h1, w2, b2, h2, C, C, T);
    conv1_sigmoid<<<dim3((T + 127) / 128, B), 128, 0, stream>>>(h2, w3, b3, atn, C, T);
}

extern "C" void kernel_launch(void* const* d_in, const int* in_sizes, int n_in,
                              void* d_out, int out_size, void* d_ws,
                              size_t ws_size, hipStream_t stream) {
    const int B = 4, D = 1024, C = 512, T = 2000;
    (void)in_sizes; (void)n_in; (void)out_size; (void)ws_size;

    const float* vfeat = (const float*)d_in[0];
    const float* ffeat = (const float*)d_in[1];
    const float* vw1 = (const float*)d_in[2];
    const float* vb1 = (const float*)d_in[3];
    const float* vw2 = (const float*)d_in[4];
    const float* vb2 = (const float*)d_in[5];
    const float* vw3 = (const float*)d_in[6];
    const float* vb3 = (const float*)d_in[7];
    const float* fw1 = (const float*)d_in[8];
    const float* fb1 = (const float*)d_in[9];
    const float* fw2 = (const float*)d_in[10];
    const float* fb2 = (const float*)d_in[11];
    const float* fw3 = (const float*)d_in[12];
    const float* fb3 = (const float*)d_in[13];

    // Output layout (return order, flat): v_atn | new_vfeat | f_atn | new_ffeat
    float* out = (float*)d_out;
    float* vatn_out = out;
    float* nvf = vatn_out + (size_t)B * T;
    float* fatn_out = nvf + (size_t)B * D * T;
    float* nff = fatn_out + (size_t)B * T;

    // Workspace layout (~97 MB)
    float* ws = (float*)d_ws;
    size_t p = 0;
    float* fusion = ws + p; p += (size_t)B * T * T;
    float* h1     = ws + p; p += (size_t)B * C * T;
    float* h2     = ws + p; p += (size_t)B * C * T;
    float* oriv   = ws + p; p += (size_t)B * T;
    float* orif   = ws + p; p += (size_t)B * T;
    float* invv   = ws + p; p += (size_t)B * T;
    float* invf   = ws + p; p += (size_t)B * T;
    float* arow   = ws + p; p += (size_t)B * T;
    float* brow   = ws + p; p += (size_t)B * T;
    float* sa     = ws + p; p += (size_t)B * T;
    float* sb     = ws + p; p += (size_t)B * T;
    float* samb   = ws + p; p += (size_t)B * T;
    float* thr    = ws + p; p += (size_t)B * T;

    // 1) original attention maps
    run_attention(vfeat, vw1, vb1, vw2, vb2, vw3, vb3, h1, h2, oriv, B, D, C, T, stream);
    run_attention(ffeat, fw1, fb1, fw2, fb2, fw3, fb3, h1, h2, orif, B, D, C, T, stream);

    // 2) per-frame inverse L2 norms
    invnorm_kernel<<<dim3((T + 127) / 128, B), 128, 0, stream>>>(vfeat, invv, D, T);
    invnorm_kernel<<<dim3((T + 127) / 128, B), 128, 0, stream>>>(ffeat, invf, D, T);

    // 3) fused, thresholded similarity matrix (WMMA)
    {
        const int NT = (T + 31) / 32;
        sim_fusion_wmma<<<dim3(NT, NT, B), 32, 0, stream>>>(
            vfeat, ffeat, invv, invf, fusion, D, T);
    }

    // 4) action / background indicators
    rowmask_kernel<<<(B * T + 255) / 256, 256, 0, stream>>>(oriv, orif, arow, brow, B * T);

    // 5) per-column top-k threshold + L1 sums
    colstats_kernel<<<dim3(T, B), 256, 0, stream>>>(
        fusion, arow, brow, thr, sa, sb, samb, T, T / 4);

    // 6) adjacency A in place
    buildA_kernel<<<dim3((unsigned)(((size_t)T * T + 255) / 256), B), 256, 0, stream>>>(
        fusion, arow, brow, thr, sa, sb, samb, T);

    // 7) feature propagation (WMMA): new = 0.5*(x + x@A)
    {
        const int NT = (T + 31) / 32;
        prop_wmma<<<dim3(NT, D / 32, B), 32, 0, stream>>>(vfeat, fusion, nvf, D, T);
        prop_wmma<<<dim3(NT, D / 32, B), 32, 0, stream>>>(ffeat, fusion, nff, D, T);
    }

    // 8) final attention maps on propagated features
    run_attention(nvf, vw1, vb1, vw2, vb2, vw3, vb3, h1, h2, vatn_out, B, D, C, T, stream);
    run_attention(nff, fw1, fb1, fw2, fb2, fw3, fb3, h1, h2, fatn_out, B, D, C, T, stream);
}
